// PackedBinaryQuadratic_25589415149864
// MI455X (gfx1250) — matrix operand order repaired and checked by reading, not compiled, not run on hardware
//
#include <hip/hip_runtime.h>
#include <hip/hip_bf16.h>
#include <stdint.h>

// ---------------------------------------------------------------------------
// Problem constants (from the reference)
// ---------------------------------------------------------------------------
#define BITS   4
#define ROWS   8
#define COLS   8
#define Y_ROW  128
#define Z_COL  128
#define KBITS  256            // bits per binary dot product
#define KBYTES 32             // 256 bits = 32 bytes = 2x uint4
#define NDIM   1024           // ROWS*Y_ROW == COLS*Z_COL == 1024
#define MDIM   16384          // BATCH*SEQ = 8*2048

typedef __attribute__((ext_vector_type(16))) __bf16 v16bf;
typedef __attribute__((ext_vector_type(8)))  float  v8f;

// fp32 -> bf16, round-to-nearest-even
__device__ __forceinline__ unsigned short f2bf(float f) {
    unsigned u = __float_as_uint(f);
    unsigned r = u + 0x7FFFu + ((u >> 16) & 1u);
    return (unsigned short)(r >> 16);
}

// ---------------------------------------------------------------------------
// Kernel 1: convert X (f32) -> bf16 workspace, 8 elements/thread
// ---------------------------------------------------------------------------
__global__ __launch_bounds__(256) void xcvt_kernel(const float* __restrict__ X,
                                                   unsigned short* __restrict__ Xbf) {
    int i = (blockIdx.x * 256 + threadIdx.x) * 8;
    const float4* p = (const float4*)(X + i);
    float4 f0 = p[0];
    float4 f1 = p[1];
    uint4 o;
    o.x = (unsigned)f2bf(f0.x) | ((unsigned)f2bf(f0.y) << 16);
    o.y = (unsigned)f2bf(f0.z) | ((unsigned)f2bf(f0.w) << 16);
    o.z = (unsigned)f2bf(f1.x) | ((unsigned)f2bf(f1.y) << 16);
    o.w = (unsigned)f2bf(f1.z) | ((unsigned)f2bf(f1.w) << 16);
    *(uint4*)(Xbf + i) = o;
}

// ---------------------------------------------------------------------------
// Kernel 2: build W[n][p] (bf16, row-major 1024x1024)
//   n = r*128 + y  (output feature dim),  p = c*128 + z (input feature dim)
//   W = sum_bit( a*popcount(Yrow & Zrow) + b*Ysum + c*Zsum ) + d
// One thread per (n,p). popcount over 32 packed bytes via v_and + v_bcnt.
// ---------------------------------------------------------------------------
__global__ __launch_bounds__(256) void wbuild_kernel(
    const unsigned char* __restrict__ Yp, const unsigned char* __restrict__ Zp,
    const int* __restrict__ Ysum, const int* __restrict__ Zsum,
    const float* __restrict__ aw, const float* __restrict__ bw,
    const float* __restrict__ cw, const float* __restrict__ dw,
    unsigned short* __restrict__ Wbf) {
    int idx = blockIdx.x * 256 + threadIdx.x;      // 0 .. 1024*1024-1
    int n = idx >> 10;                             // row of W (feature out)
    int p = idx & (NDIM - 1);                      // col of W (feature in)
    int r = n >> 7, y = n & 127;
    int c = p >> 7, z = p & 127;

    float acc = dw[0];
#pragma unroll
    for (int bit = 0; bit < BITS; ++bit) {
        long ybase = ((((long)bit * ROWS + r) * COLS + c) * Y_ROW + y) * KBYTES;
        long zbase = ((((long)bit * ROWS + r) * COLS + c) * Z_COL + z) * KBYTES;
        uint4 y0 = *(const uint4*)(Yp + ybase);
        uint4 y1 = *(const uint4*)(Yp + ybase + 16);
        uint4 z0 = *(const uint4*)(Zp + zbase);
        uint4 z1 = *(const uint4*)(Zp + zbase + 16);
        int pop = __popc(y0.x & z0.x) + __popc(y0.y & z0.y) +
                  __popc(y0.z & z0.z) + __popc(y0.w & z0.w) +
                  __popc(y1.x & z1.x) + __popc(y1.y & z1.y) +
                  __popc(y1.z & z1.z) + __popc(y1.w & z1.w);
        int ys = Ysum[(((bit * ROWS + r) * COLS + c) * Y_ROW) + y];
        int zs = Zsum[(((bit * ROWS + r) * COLS + c) * Z_COL) + z];
        acc += aw[bit] * (float)pop + bw[bit] * (float)ys + cw[bit] * (float)zs;
    }
    Wbf[(long)n * NDIM + p] = f2bf(acc);
}

// ---------------------------------------------------------------------------
// Kernel 3: out[M][N] = Xbf[M][K] * Wbf[N][K]^T + bias[N]   (K = N = 1024)
// WMMA bf16 16x16x32, f32 accumulate. 8 waves/block, wave tile 64x64,
// block tile 128(M) x 256(N). Fragments loaded per the ISA VGPR layouts:
//   A 16x32 (16-bit): lane m=l16, half=lane>>4; elems 0-7 <- K = 8*half..+7,
//                     elems 8-15 <- K = 16+8*half..+7         (two b128 loads)
//   B 32x16 (16-bit): lane n=l16; elems 0-7 <- K = 8*half..+7, elems 8-15 <- K = 16+8*half..+7
//                     (two b128 loads of one W row)
//   C/D 16x16 f32:    vgpr v: lanes0-15 -> M=v, lanes16-31 -> M=8+v; N=l16
// ---------------------------------------------------------------------------
union AFrag { v16bf v; uint4 u[2]; };

__global__ __launch_bounds__(256) void gemm_wmma_kernel(
    const unsigned short* __restrict__ Xbf,
    const unsigned short* __restrict__ Wbf,
    const float* __restrict__ bias,
    float* __restrict__ out) {
    const int lane = threadIdx.x & 31;
    const int wave = threadIdx.x >> 5;
    const int l16  = lane & 15;
    const int half = lane >> 4;

    const int mwave = blockIdx.x * 128 + (wave >> 2) * 64;   // M base of wave
    const int nwave = blockIdx.y * 256 + (wave & 3) * 64;    // N base of wave

    v8f acc[4][4] = {};

    for (int k = 0; k < NDIM; k += 32) {
        AFrag a[4], b[4];
#pragma unroll
        for (int i = 0; i < 4; ++i) {
            const unsigned short* p =
                Xbf + (long)(mwave + i * 16 + l16) * NDIM + k + half * 8;
            a[i].u[0] = *(const uint4*)p;          // K = k+8*half .. +7
            a[i].u[1] = *(const uint4*)(p + 16);   // K = k+16+8*half .. +7
        }
#pragma unroll
        for (int j = 0; j < 4; ++j) {
            const unsigned short* p =
                Wbf + (long)(nwave + j * 16 + l16) * NDIM + k + half * 8;
            b[j].u[0] = *(const uint4*)p;          // K = k+8*half .. +7
            b[j].u[1] = *(const uint4*)(p + 16);   // K = k+16+8*half .. +7
        }
        if (k + 32 < NDIM) {                       // gfx1250 global_prefetch_b8
            __builtin_prefetch(Xbf + (long)(mwave + l16) * NDIM + k + 32, 0, 1);
            __builtin_prefetch(Wbf + (long)(nwave + l16) * NDIM + k + 32, 0, 1);
        }
#pragma unroll
        for (int i = 0; i < 4; ++i)
#pragma unroll
            for (int j = 0; j < 4; ++j)
                acc[i][j] = __builtin_amdgcn_wmma_f32_16x16x32_bf16(
                    /*neg_a=*/false, a[i].v, /*neg_b=*/false, b[j].v,
                    /*c_mod=*/(short)0, acc[i][j],
                    /*reuse_a=*/false, /*reuse_b=*/false);
    }

    // Epilogue: D layout -> global, + bias over feature (N) dim
#pragma unroll
    for (int j = 0; j < 4; ++j) {
        int n = nwave + j * 16 + l16;
        float bv = bias[n];
#pragma unroll
        for (int i = 0; i < 4; ++i) {
#pragma unroll
            for (int rr = 0; rr < 8; ++rr) {
                int m = mwave + i * 16 + half * 8 + rr;
                out[(long)m * NDIM + n] = acc[i][j][rr] + bv;
            }
        }
    }
}

// ---------------------------------------------------------------------------
// Launch: inputs are (X, Y_packed, Z_packed, Y_sum, Z_sum, a, b, c, d, bias)
// Workspace: [0, 32MB) = Xbf, [32MB, 34MB) = Wbf
// ---------------------------------------------------------------------------
extern "C" void kernel_launch(void* const* d_in, const int* in_sizes, int n_in,
                              void* d_out, int out_size, void* d_ws, size_t ws_size,
                              hipStream_t stream) {
    const float*         X    = (const float*)d_in[0];
    const unsigned char* Yp   = (const unsigned char*)d_in[1];
    const unsigned char* Zp   = (const unsigned char*)d_in[2];
    const int*           Ysum = (const int*)d_in[3];
    const int*           Zsum = (const int*)d_in[4];
    const float*         aw   = (const float*)d_in[5];
    const float*         bw   = (const float*)d_in[6];
    const float*         cw   = (const float*)d_in[7];
    const float*         dw   = (const float*)d_in[8];
    const float*         bias = (const float*)d_in[9];
    float*               out  = (float*)d_out;

    unsigned short* Xbf = (unsigned short*)d_ws;                       // 32 MB
    unsigned short* Wbf = (unsigned short*)((char*)d_ws +
                              (size_t)MDIM * NDIM * sizeof(unsigned short)); // 2 MB

    // 1) X -> bf16   (16384*1024 elems, 8 per thread)
    xcvt_kernel<<<(MDIM * NDIM) / (256 * 8), 256, 0, stream>>>(X, Xbf);

    // 2) Build W (bf16 1024x1024)
    wbuild_kernel<<<(NDIM * NDIM) / 256, 256, 0, stream>>>(
        Yp, Zp, Ysum, Zsum, aw, bw, cw, dw, Wbf);

    // 3) WMMA GEMM + bias
    dim3 grid(MDIM / 128, NDIM / 256);
    gemm_wmma_kernel<<<grid, 256, 0, stream>>>(Xbf, Wbf, bias, out);
}
